// WAP_42880953483311
// MI455X (gfx1250) — compile-verified, matrix operand order checked by reading, not run on hardware
//
#include <hip/hip_runtime.h>
#include <math.h>

// ---------------------------------------------------------------------------
// MI455X (gfx1250, wave32).
// Encoder convs: implicit-GEMM, V_WMMA_F32_16X16X32_BF16 (bf16 in, f32 acc).
//   - weights pre-swizzled into WMMA fragment order (global) once,
//     then staged to LDS with GLOBAL_LOAD_ASYNC_TO_LDS_B128 (ASYNCcnt path)
//   - im2col A gather fully compile-time-shifted (H,W,C template params)
// Decoder: single-workgroup fp32 loop, wave32 butterfly reductions.
// ---------------------------------------------------------------------------

typedef __attribute__((ext_vector_type(16))) __bf16 v16bf;
typedef __attribute__((ext_vector_type(8)))  float  v8f;
typedef __attribute__((ext_vector_type(4)))  int    v4i;

typedef __attribute__((address_space(1))) v4i* gas_v4i_ptr;
typedef __attribute__((address_space(3))) v4i* lds_v4i_ptr;

#define NUM_TOKEN 112
#define MAX_LEN 32
#define BATCH 4
#define DDIM 128
#define AHW 512   // 16*32 attention grid

#if __has_builtin(__builtin_amdgcn_global_load_async_to_lds_b128)
#define USE_ASYNC_LDS 1
#else
#define USE_ASYNC_LDS 0
#endif

__device__ __forceinline__ void wait_asynccnt0() {
#if __has_builtin(__builtin_amdgcn_s_wait_asynccnt)
    __builtin_amdgcn_s_wait_asynccnt(0);
#else
    asm volatile("s_wait_asynccnt 0" ::: "memory");
#endif
}

__device__ __forceinline__ float wave_add(float v) {
#pragma unroll
    for (int o = 16; o > 0; o >>= 1) v += __shfl_xor(v, o, 32);
    return v;
}
__device__ __forceinline__ float wave_max(float v) {
#pragma unroll
    for (int o = 16; o > 0; o >>= 1) v = fmaxf(v, __shfl_xor(v, o, 32));
    return v;
}

// ---------------------------------------------------------------------------
// Weight repack: w[o][c][kh][kw] fp32 -> bf16, already in WMMA B-fragment
// order, chunked along K (K = (kh*3+kw)*C + c, 32 K per chunk, zero padded).
// ---------------------------------------------------------------------------
__global__ void cvt_w_sw(const float* __restrict__ src, __bf16* __restrict__ dst,
                         int C, int O, int nch) {
    int NT = O / 16;
    int per = NT * 512;
    int total = nch * per;
    int K = 9 * C;
    for (int i = blockIdx.x * blockDim.x + threadIdx.x; i < total;
         i += gridDim.x * blockDim.x) {
        int ck = i / per;
        int r  = i - ck * per;
        int j = r >> 9, l = (r >> 4) & 31, e = r & 15;
        int n = j * 16 + (l & 15);
        int k = ck * 32 + e + 8 * ((e >> 3) + (l >> 4));
        float v = 0.0f;
        if (k < K) {
            int p = k / C, c = k - p * C;   // tap, input channel
            v = src[(n * C + c) * 9 + p];
        }
        dst[i] = (__bf16)v;
    }
}

// ---------------------------------------------------------------------------
// Implicit-GEMM conv3x3 (SAME) + bias + ReLU (see round-3 comments).
// ---------------------------------------------------------------------------
template <int C, int O, int H, int W, bool INF32>
__global__ __launch_bounds__(256) void conv_wmma(
    const __bf16* __restrict__ in_bf, const float* __restrict__ in_f,
    const __bf16* __restrict__ wgt_sw, const float* __restrict__ bias,
    __bf16* __restrict__ out) {
    constexpr int K   = 9 * C;
    constexpr int NT  = O / 16;
    constexpr int HW  = H * W;            // power of two
    constexpr int NCH = (K + 31) / 32;

    __shared__ __align__(32) __bf16 sA[8 * 512];
    __shared__ __align__(32) __bf16 sB[NT * 512];

    const int tid  = threadIdx.x;
    const int wave = tid >> 5;
    const int lane = tid & 31;
    const int m0   = blockIdx.x * 128;

    v8f acc[NT] = {};

    for (int ck = 0; ck < NCH; ++ck) {
        const int kc = ck * 32;

        const __bf16* bsrc = wgt_sw + (size_t)ck * (NT * 512);
#if USE_ASYNC_LDS
        for (int t = tid; t < NT * 64; t += 256) {
            __builtin_amdgcn_global_load_async_to_lds_b128(
                (gas_v4i_ptr)(bsrc + t * 8),
                (lds_v4i_ptr)(sB + t * 8),
                0, 0);
        }
#else
        {
            const uint4* gs = (const uint4*)bsrc;
            uint4* ss = (uint4*)sB;
            for (int t = tid; t < NT * 64; t += 256) ss[t] = gs[t];
        }
#endif

        {
            const int t = tid >> 5;
            const int l = tid & 31;
            const int m = m0 + t * 16 + (l & 15);
            const int half = l >> 4;
            const int b   = m / HW;
            const int rem = m & (HW - 1);
            const int y   = rem / W;
            const int x   = rem & (W - 1);
#pragma unroll
            for (int e = 0; e < 16; ++e) {
                int k = kc + e + 8 * ((e >> 3) + half);
                float v = 0.0f;
                if (k < K) {
                    int p = k / C;
                    int c = k & (C - 1);
                    int yy = y + p / 3 - 1;
                    int xx = x + p % 3 - 1;
                    if ((unsigned)yy < (unsigned)H && (unsigned)xx < (unsigned)W) {
                        int off = ((b * C + c) * H + yy) * W + xx;
                        v = INF32 ? in_f[off] : (float)in_bf[off];
                    }
                }
                sA[(t * 32 + l) * 16 + e] = (__bf16)v;
            }
        }
#if USE_ASYNC_LDS
        wait_asynccnt0();
#endif
        __syncthreads();

        v16bf afrag = *(const v16bf*)(sA + wave * 512 + lane * 16);
#pragma unroll
        for (int j = 0; j < NT; ++j) {
            v16bf bfrag = *(const v16bf*)(sB + j * 512 + lane * 16);
            acc[j] = __builtin_amdgcn_wmma_f32_16x16x32_bf16(
                false, afrag, false, bfrag, (short)0, acc[j], false, false);
        }
        __syncthreads();
    }

    const int mBase = m0 + wave * 16 + 8 * (lane >> 4);
#pragma unroll
    for (int j = 0; j < NT; ++j) {
        int n = j * 16 + (lane & 15);
        float bv = bias[n];
#pragma unroll
        for (int r = 0; r < 8; ++r) {
            int m = mBase + r;
            int b = m / HW;
            int rem = m & (HW - 1);
            int y = rem / W, x = rem & (W - 1);
            float v = acc[j][r] + bv;
            v = v > 0.0f ? v : 0.0f;
            out[((b * O + n) * H + y) * W + x] = (__bf16)v;
        }
    }
}

// ---------------------------------------------------------------------------
// relu(maxpool2x2) on bf16; optionally also emit fp32 (final feature map).
// ---------------------------------------------------------------------------
__global__ void pool_relu(const __bf16* __restrict__ in, __bf16* __restrict__ out,
                          float* __restrict__ outf, int Cc, int H, int W) {
    int Ho = H >> 1, Wo = W >> 1;
    int n = BATCH * Cc * Ho * Wo;
    for (int idx = blockIdx.x * blockDim.x + threadIdx.x; idx < n;
         idx += gridDim.x * blockDim.x) {
        int x = idx % Wo;
        int t = idx / Wo;
        int y = t % Ho; t /= Ho;
        int c = t % Cc;
        int b = t / Cc;
        const __bf16* p = &in[((b * Cc + c) * H + 2 * y) * W + 2 * x];
        float v0 = (float)p[0], v1 = (float)p[1];
        float v2 = (float)p[W], v3 = (float)p[W + 1];
        float v = fmaxf(fmaxf(v0, v1), fmaxf(v2, v3));
        v = fmaxf(v, 0.0f);
        out[idx] = (__bf16)v;
        if (outf) outf[idx] = v;
    }
}

// ---------------------------------------------------------------------------
// Decoder: 31 sequential attention+GRU steps, one 512-thread workgroup
// (16 wave32s; each batch row b owns waves 4b..4b+3).  All softmax / dot
// reductions are wave butterfly (__shfl_xor) + 4-partial LDS combine.
// F: [4][128][512] fp32 feature map.  out: [4][32][112] fp32.
// ---------------------------------------------------------------------------
__global__ __launch_bounds__(512) void decoder_kernel(
    const float* __restrict__ F,   const float* __restrict__ h0,
    const float* __restrict__ Wih, const float* __restrict__ Whh,
    const float* __restrict__ bih, const float* __restrict__ bhh,
    const float* __restrict__ Wp,  const float* __restrict__ bp,
    const float* __restrict__ Wh,  const float* __restrict__ bh,
    const float* __restrict__ Wa,  const float* __restrict__ ba,
    float* __restrict__ out) {
    const int tid  = threadIdx.x;
    const int b    = tid >> 7;        // batch row owned by this thread's group
    const int sub  = tid & 127;       // index within the 4-wave group
    const int wv   = tid >> 5;        // wave id 0..15
    __shared__ float sh_h[BATCH * DDIM];
    __shared__ float sh_ctx[BATCH * DDIM];
    __shared__ float sh_alpha[BATCH * AHW];
    __shared__ float sh_froma[BATCH * AHW];
    __shared__ float sh_g[BATCH * 768];          // [b][0:384)=gi, [384:768)=gh
    __shared__ float sh_red[16];                 // one slot per wave

    for (int i = tid; i < BATCH * DDIM; i += 512) sh_h[i] = h0[i];
    for (int i = tid; i < BATCH * AHW; i += 512) sh_alpha[i] = 0.0f;
    for (int i = tid; i < BATCH * NUM_TOKEN; i += 512) {
        int bb = i / NUM_TOKEN, t = i - bb * NUM_TOKEN;
        out[(bb * MAX_LEN + 0) * NUM_TOKEN + t] = 0.0f;
    }
    for (int i = tid; i < BATCH * AHW; i += 512) {
        int bb = i >> 9, hw = i & 511;
        float s = ba[0];
        for (int c = 0; c < DDIM; ++c) s += F[(bb * DDIM + c) * AHW + hw] * Wa[c];
        sh_froma[i] = s;
    }
    __syncthreads();

    for (int s = 0; s < MAX_LEN - 1; ++s) {
        // ctx[b][c] = sum_hw F[b][c][hw] * alpha[b][hw]
        for (int i = tid; i < BATCH * DDIM; i += 512) {
            int bb = i >> 7, c = i & 127;
            const float* fp = &F[(bb * DDIM + c) * AHW];
            const float* ap = &sh_alpha[bb * AHW];
            float acc = 0.0f;
            for (int hw = 0; hw < AHW; ++hw) acc += fp[hw] * ap[hw];
            sh_ctx[i] = acc;
        }
        __syncthreads();
        // GRU gate pre-activations
        for (int i = tid; i < BATCH * 384; i += 512) {
            int bb = i / 384, g = i - bb * 384;
            float gi = bih[g], gh = bhh[g];
            const float* wi = &Wih[g * DDIM];
            const float* wh = &Whh[g * DDIM];
            const float* cx = &sh_ctx[bb * DDIM];
            const float* hh = &sh_h[bb * DDIM];
            for (int d = 0; d < DDIM; ++d) { gi += wi[d] * cx[d]; gh += wh[d] * hh[d]; }
            sh_g[bb * 768 + g] = gi;
            sh_g[bb * 768 + 384 + g] = gh;
        }
        __syncthreads();
        // gates + state update (elementwise, in place)
        for (int i = tid; i < BATCH * DDIM; i += 512) {
            int bb = i >> 7, d = i & 127;
            float ir = sh_g[bb * 768 + d];
            float iz = sh_g[bb * 768 + 128 + d];
            float in = sh_g[bb * 768 + 256 + d];
            float hr = sh_g[bb * 768 + 384 + d];
            float hz = sh_g[bb * 768 + 512 + d];
            float hn = sh_g[bb * 768 + 640 + d];
            float r = 1.0f / (1.0f + expf(-(ir + hr)));
            float z = 1.0f / (1.0f + expf(-(iz + hz)));
            float n = tanhf(in + r * hn);
            sh_h[i] = (1.0f - z) * n + z * sh_h[i];
        }
        __syncthreads();

        // ---- token logits + log-softmax (per-b over 112, wave-parallel) ----
        float lv = -1e30f;
        if (sub < NUM_TOKEN) {
            float p = bp[sub];
            const float* wp = &Wp[sub * DDIM];
            const float* hh = &sh_h[b * DDIM];
            for (int d = 0; d < DDIM; ++d) p += wp[d] * hh[d];
            lv = p;
        }
        float wm = wave_max(lv);
        if ((tid & 31) == 0) sh_red[wv] = wm;
        __syncthreads();
        float bm = fmaxf(fmaxf(sh_red[b * 4 + 0], sh_red[b * 4 + 1]),
                         fmaxf(sh_red[b * 4 + 2], sh_red[b * 4 + 3]));
        float ev = (sub < NUM_TOKEN) ? expf(lv - bm) : 0.0f;
        float ws = wave_add(ev);
        __syncthreads();
        if ((tid & 31) == 0) sh_red[wv] = ws;
        __syncthreads();
        float bs = sh_red[b * 4 + 0] + sh_red[b * 4 + 1] +
                   sh_red[b * 4 + 2] + sh_red[b * 4 + 3];
        float lse = bm + logf(bs);
        if (sub < NUM_TOKEN)
            out[(b * MAX_LEN + s + 1) * NUM_TOKEN + sub] = lv - lse;

        // ---- from_h[b] = bh + Wh . h[b]  (128-wide parallel dot) ----
        float fv = Wh[sub] * sh_h[b * DDIM + sub];
        float fw = wave_add(fv);
        __syncthreads();
        if ((tid & 31) == 0) sh_red[wv] = fw;
        __syncthreads();
        float fh = bh[0] + sh_red[b * 4 + 0] + sh_red[b * 4 + 1] +
                   sh_red[b * 4 + 2] + sh_red[b * 4 + 3];

        // ---- attention: a = tanh(fh + from_a), softmax over 512 per b ----
        float sc[4];
#pragma unroll
        for (int j = 0; j < 4; ++j)
            sc[j] = tanhf(fh + sh_froma[b * AHW + sub + 128 * j]);
        float am = fmaxf(fmaxf(sc[0], sc[1]), fmaxf(sc[2], sc[3]));
        am = wave_max(am);
        __syncthreads();
        if ((tid & 31) == 0) sh_red[wv] = am;
        __syncthreads();
        float abm = fmaxf(fmaxf(sh_red[b * 4 + 0], sh_red[b * 4 + 1]),
                          fmaxf(sh_red[b * 4 + 2], sh_red[b * 4 + 3]));
        float es = 0.0f;
#pragma unroll
        for (int j = 0; j < 4; ++j) { sc[j] = expf(sc[j] - abm); es += sc[j]; }
        es = wave_add(es);
        __syncthreads();
        if ((tid & 31) == 0) sh_red[wv] = es;
        __syncthreads();
        float abs_ = sh_red[b * 4 + 0] + sh_red[b * 4 + 1] +
                     sh_red[b * 4 + 2] + sh_red[b * 4 + 3];
        float inv = 1.0f / abs_;
#pragma unroll
        for (int j = 0; j < 4; ++j)
            sh_alpha[b * AHW + sub + 128 * j] = sc[j] * inv;
        __syncthreads();
    }
}

// ---------------------------------------------------------------------------
// Host side
// ---------------------------------------------------------------------------
extern "C" void kernel_launch(void* const* d_in, const int* in_sizes, int n_in,
                              void* d_out, int out_size, void* d_ws, size_t ws_size,
                              hipStream_t stream) {
    (void)in_sizes; (void)n_in; (void)out_size; (void)ws_size;
    const float* x  = (const float*)d_in[0];
    const float* h0 = (const float*)d_in[1];
    const float* Wih = (const float*)d_in[34];
    const float* Whh = (const float*)d_in[35];
    const float* bih = (const float*)d_in[36];
    const float* bhh = (const float*)d_in[37];
    const float* Wp  = (const float*)d_in[38];
    const float* bp  = (const float*)d_in[39];
    const float* Wh  = (const float*)d_in[40];
    const float* bh  = (const float*)d_in[41];
    const float* Wa  = (const float*)d_in[42];
    const float* ba  = (const float*)d_in[43];
    float* out = (float*)d_out;

    static const int OS[16]   = {32,32,32,32, 64,64,64,64, 64,64,64,64, 128,128,128,128};
    static const int CSin[16] = {1,32,32,32, 32,64,64,64, 64,64,64,64, 64,128,128,128};
    static const int KIND[16] = {0,1,1,1, 2,3,3,3, 4,4,4,4, 5,6,6,6};

    const size_t ACT_ELEMS = (size_t)4 * 32 * 256 * 512;
    __bf16* bufA = (__bf16*)d_ws;
    __bf16* bufB = bufA + ACT_ELEMS;
    __bf16* wreg = (__bf16*)((char*)d_ws + 2 * ACT_ELEMS * sizeof(__bf16));
    float*  Fres = (float*)((char*)d_ws + 2 * ACT_ELEMS * sizeof(__bf16) + (2u << 20));

    size_t woffs[16]; size_t wo = 0;
    for (int i = 0; i < 16; ++i) {
        woffs[i] = wo;
        int nch = (9 * CSin[i] + 31) / 32;
        int total = nch * (OS[i] / 16) * 512;
        const float* wsrc = (const float*)d_in[2 + 2 * i];
        cvt_w_sw<<<(total + 255) / 256, 256, 0, stream>>>(wsrc, wreg + wo,
                                                          CSin[i], OS[i], nch);
        wo += (size_t)total;
    }

    int H = 256, W = 512;
    __bf16* cur = nullptr;
    for (int i = 0; i < 16; ++i) {
        __bf16* dst = (cur == bufA) ? bufB : bufA;
        const float* bias = (const float*)d_in[3 + 2 * i];
        const __bf16* w = wreg + woffs[i];
        dim3 g(BATCH * H * W / 128), blk(256);
        switch (KIND[i]) {
        case 0: conv_wmma<1,  32, 256,512, true ><<<g,blk,0,stream>>>(nullptr, x, w, bias, dst); break;
        case 1: conv_wmma<32, 32, 256,512, false><<<g,blk,0,stream>>>(cur, nullptr, w, bias, dst); break;
        case 2: conv_wmma<32, 64, 128,256, false><<<g,blk,0,stream>>>(cur, nullptr, w, bias, dst); break;
        case 3: conv_wmma<64, 64, 128,256, false><<<g,blk,0,stream>>>(cur, nullptr, w, bias, dst); break;
        case 4: conv_wmma<64, 64,  64,128, false><<<g,blk,0,stream>>>(cur, nullptr, w, bias, dst); break;
        case 5: conv_wmma<64,128,  32, 64, false><<<g,blk,0,stream>>>(cur, nullptr, w, bias, dst); break;
        default:conv_wmma<128,128, 32, 64, false><<<g,blk,0,stream>>>(cur, nullptr, w, bias, dst); break;
        }
        cur = dst;
        if (i == 3 || i == 7 || i == 11 || i == 15) {
            __bf16* pdst = (cur == bufA) ? bufB : bufA;
            float* f32dst = (i == 15) ? Fres : nullptr;
            int n = BATCH * OS[i] * (H / 2) * (W / 2);
            int blocks = (n + 255) / 256;
            if (blocks > 4096) blocks = 4096;
            pool_relu<<<blocks, 256, 0, stream>>>(cur, pdst, f32dst, OS[i], H, W);
            cur = pdst;
            H >>= 1; W >>= 1;
        }
    }

    decoder_kernel<<<1, 512, 0, stream>>>(Fres, h0, Wih, Whh, bih, bhh,
                                          Wp, bp, Wh, bh, Wa, ba, out);
}
